// RelativeMultiheadAttention_53034256171183
// MI455X (gfx1250) — matrix operand order repaired and checked by reading, not compiled, not run on hardware
//
#include <hip/hip_runtime.h>

// ---------------------------------------------------------------------------
// XLNet-style relative multi-head attention for gfx1250 (MI455X, wave32).
// - All matmuls: v_wmma_f32_16x16x32_bf16 (f32 accumulate).
// - Attention operands (K, Kr, V^T) are staged into LDS by the Tensor Data
//   Mover (tensor_load_to_lds + s_wait_tensorcnt), double-buffered, shared by
//   the 4 waves of a workgroup (same (b,n), 4 consecutive 16-row q blocks).
// - rel_shift done in-register via skewed WMMA windows + lane shuffles.
// - Online (flash-style) softmax: no [q,klen]/[q,rlen] score tensors in HBM.
// ---------------------------------------------------------------------------

#define QLEN 1024
#define MLEN 1024
#define KLEN 2048
#define RLEN 3072
#define BSZ  2
#define NH   12
#define DH   64
#define DM   768
#define ATT_SCALE 0.125f   // 1/sqrt(64)

typedef __bf16        bf16x16 __attribute__((ext_vector_type(16)));
typedef float         f32x8   __attribute__((ext_vector_type(8)));
typedef unsigned int  u32x4   __attribute__((ext_vector_type(4)));
typedef int           i32x4   __attribute__((ext_vector_type(4)));
typedef int           i32x8   __attribute__((ext_vector_type(8)));

union BfPack { bf16x16 v; u32x4 q[2]; };

__device__ __forceinline__ unsigned short f2bf(float f) {
  unsigned int u = __float_as_uint(f);
  u += 0x7FFFu + ((u >> 16) & 1u);          // round-to-nearest-even
  return (unsigned short)(u >> 16);
}

__device__ __forceinline__ f32x8 wmma_bf16(const BfPack& a, const BfPack& b, f32x8 c) {
  return __builtin_amdgcn_wmma_f32_16x16x32_bf16(false, a.v, false, b.v,
                                                 (short)0, c, false, false);
}

// Low 32 bits of a generic pointer to __shared__ = LDS byte offset.
__device__ __forceinline__ unsigned ldsoff(const void* p) {
  return (unsigned)(unsigned long long)p;
}

// ---- Tensor Data Mover: 2D tile load, bf16 (data_size = 2B) ----------------
// D# per CDNA5 ISA: group0 = {count=1, lds_addr, global_addr(57b), type=2}
// group1 = {data_size=1(2B), tensor_dim0/1 == tile_dim0/1, dim0_stride}.
// Tensor dims == tile dims so no OOB clamping path is exercised.
// This toolchain's builtin takes 6 args (g0, g1, g2, g3, extra, cpol).
__device__ __forceinline__ void tdm_load_2d(unsigned lds_addr, const void* gptr,
                                            int tile0, int tile1,
                                            long long stride_elems) {
  unsigned long long ga = (unsigned long long)gptr;
  u32x4 g0;
  g0[0] = 1u;                                            // count=1, user mode
  g0[1] = lds_addr;                                      // LDS byte address
  g0[2] = (unsigned)(ga & 0xFFFFFFFFu);                  // global_addr[31:0]
  g0[3] = (unsigned)((ga >> 32) & 0x01FFFFFFu) | (2u << 30);  // [56:32] | type=2
  i32x8 g1;
  g1[0] = 0x00010000;                                    // data_size=1 -> 2 bytes
  g1[1] = (tile0 & 0xFFFF) << 16;                        // tensor_dim0 lo16
  g1[2] = ((tile0 >> 16) & 0xFFFF) | ((tile1 & 0xFFFF) << 16); // dim0 hi | dim1 lo
  g1[3] = ((tile1 >> 16) & 0xFFFF) | ((tile0 & 0xFFFF) << 16); // dim1 hi | tile_dim0
  g1[4] = tile1 & 0xFFFF;                                // tile_dim1 (tile_dim2=0)
  g1[5] = (int)(stride_elems & 0xFFFFFFFFll);            // dim0_stride lo32
  g1[6] = (int)((stride_elems >> 32) & 0xFFFFll);        // dim0_stride hi16
  g1[7] = 0;
  i32x4 gz4 = {0, 0, 0, 0};
  i32x8 gz8 = {0, 0, 0, 0, 0, 0, 0, 0};
  __builtin_amdgcn_tensor_load_to_lds(g0, g1, gz4, gz4, gz8, 0);
}

// ------------------------------ pack / convert ------------------------------

__global__ void k_cat_cvt(const float* __restrict__ mems, const float* __restrict__ h,
                          unsigned short* __restrict__ dst) {
  int i = blockIdx.x * blockDim.x + threadIdx.x;
  const int half = MLEN * BSZ * DM;
  if (i >= 2 * half) return;
  float v = (i < half) ? mems[i] : h[i - half];
  dst[i] = f2bf(v);
}

__global__ void k_cvt(const float* __restrict__ s, unsigned short* __restrict__ d, int n) {
  int i = blockIdx.x * blockDim.x + threadIdx.x;
  if (i < n) d[i] = f2bf(s[i]);
}

// wt[(n*64+d), hm] = w[hm, n, d]   (B^T for the projection GEMMs)
__global__ void k_wtrans(const float* __restrict__ w, unsigned short* __restrict__ wt) {
  int i = blockIdx.x * blockDim.x + threadIdx.x;
  if (i >= DM * DM) return;
  int c = i / DM, k = i % DM;
  wt[i] = f2bf(w[k * DM + c]);
}

__global__ void k_addbias_cvt(const float* __restrict__ s, const float* __restrict__ bias,
                              unsigned short* __restrict__ d, int n) {
  int i = blockIdx.x * blockDim.x + threadIdx.x;
  if (i >= n) return;
  d[i] = f2bf(s[i] + bias[i % DM]);
}

// vT[((b*12+n)*64+d), j] = v[(j*2+b)*768 + n*64 + d]
__global__ void k_vtrans(const float* __restrict__ vf, unsigned short* __restrict__ vT) {
  int i = blockIdx.x * blockDim.x + threadIdx.x;
  if (i >= BSZ * NH * DH * KLEN) return;
  int j  = i & (KLEN - 1);
  int d  = (i >> 11) & (DH - 1);
  int bn = i >> 17;
  int b = bn / NH, n = bn % NH;
  vT[i] = f2bf(vf[(size_t)(j * BSZ + b) * DM + n * DH + d]);
}

// e[(row*12+n)*2 + s] = (q[row,n,:] + r_s_bias[n,:]) . seg_embed[s,n,:]
__global__ void k_eseg(const float* __restrict__ qf, const float* __restrict__ rsb,
                       const float* __restrict__ seg, float* __restrict__ e) {
  int idx = blockIdx.x * blockDim.x + threadIdx.x;
  if (idx >= QLEN * BSZ * NH) return;
  int n = idx % NH, row = idx / NH;
  float e0 = 0.f, e1 = 0.f;
  const float* qp = qf + (size_t)row * DM + n * DH;
  const float* bp = rsb + n * DH;
  const float* s0 = seg + n * DH;
  const float* s1 = seg + DM + n * DH;
  for (int d = 0; d < DH; ++d) {
    float qv = qp[d] + bp[d];
    e0 += qv * s0[d];
    e1 += qv * s1[d];
  }
  e[idx * 2 + 0] = e0;
  e[idx * 2 + 1] = e1;
}

// ------------------------------ bf16 WMMA GEMM ------------------------------
// C[M,N] f32 = A[M,K] bf16 x Bt[N,K] bf16.  One wave computes a 32x64 tile
// (2 A-frags x 4 B-frags -> 8 WMMA per 32-wide K step) for operand reuse.
__global__ __launch_bounds__(128)
void k_gemm_bf16(const unsigned short* __restrict__ A,
                 const unsigned short* __restrict__ Bt,
                 float* __restrict__ C, int M, int N, int K) {
  int lane = threadIdx.x & 31;
  int w    = threadIdx.x >> 5;
  int wg   = blockIdx.x * 4 + w;
  int tilesN = N >> 6;
  int tm = wg / tilesN, tn = wg % tilesN;
  if (tm * 32 >= M) return;
  int lcol = lane & 15, lh = lane >> 4;
  int m0 = tm * 32, n0 = tn * 64;

  f32x8 zero = {};
  f32x8 acc[2][4];
#pragma unroll
  for (int i = 0; i < 2; ++i)
#pragma unroll
    for (int t = 0; t < 4; ++t) acc[i][t] = zero;

  const unsigned short* a0 = A + (size_t)(m0 + lcol) * K;
  const unsigned short* a1 = A + (size_t)(m0 + 16 + lcol) * K;
  const unsigned short* br = Bt + (size_t)(n0 + lcol) * K;
  for (int k0 = 0; k0 < K; k0 += 32) {
    BfPack af[2], bfr[4];
    const unsigned short* ap0 = a0 + k0 + 8 * lh;
    af[0].q[0] = *(const u32x4*)ap0;  af[0].q[1] = *(const u32x4*)(ap0 + 16);
    const unsigned short* ap1 = a1 + k0 + 8 * lh;
    af[1].q[0] = *(const u32x4*)ap1;  af[1].q[1] = *(const u32x4*)(ap1 + 16);
#pragma unroll
    for (int t = 0; t < 4; ++t) {
      const unsigned short* bp = br + (size_t)(16 * t) * K + k0 + 8 * lh;
      bfr[t].q[0] = *(const u32x4*)bp;  bfr[t].q[1] = *(const u32x4*)(bp + 16);
    }
#pragma unroll
    for (int i = 0; i < 2; ++i)
#pragma unroll
      for (int t = 0; t < 4; ++t) acc[i][t] = wmma_bf16(af[i], bfr[t], acc[i][t]);
  }
#pragma unroll
  for (int i = 0; i < 2; ++i)
#pragma unroll
    for (int t = 0; t < 4; ++t)
#pragma unroll
      for (int v = 0; v < 8; ++v) {
        int row = m0 + 16 * i + v + 8 * lh;
        C[(size_t)row * N + n0 + 16 * t + lcol] = acc[i][t][v];
      }
}

// --------------------------- fused attention core ---------------------------
// Block = 4 waves on one (b,n); wave w owns query rows [i0base+16w, +16).
// Per 32-key step, TDM stages shared tiles into double-buffered LDS:
//   wave0: K   [32 j x 64 d]      wave1: V^T [64 d x 32 j]
//   wave2/3: Kr skew window, 96 rows x 64 d (union of the 4 waves' windows).
__global__ __launch_bounds__(128)
void k_xl_attention(const unsigned short* __restrict__ qwbf,
                    const unsigned short* __restrict__ qrbf,
                    const unsigned short* __restrict__ kbf,
                    const unsigned short* __restrict__ krbf,
                    const unsigned short* __restrict__ vTbf,
                    const float* __restrict__ e2,
                    const unsigned char* __restrict__ segmat,
                    unsigned short* __restrict__ attnbf) {
  __shared__ unsigned short kbuf[2][32][64];     //  8 KB
  __shared__ unsigned short vbuf[2][64][32];     //  8 KB
  __shared__ unsigned short krbuf[2][96][64];    // 24 KB
  __shared__ unsigned short plds[4][16][32];     //  4 KB

  int lane = threadIdx.x & 31;
  int w    = threadIdx.x >> 5;                   // wave in block: 0..3
  int blk  = blockIdx.x;                         // 0..383
  int gb = blk & 15;
  int n  = (blk >> 4) % NH;
  int b  = blk / (16 * NH);
  int i0base = gb * 64;
  int i0 = i0base + w * 16;
  int lcol = lane & 15, lh = lane >> 4;

  // Resident A operands: q + r_w_bias and q + r_r_bias (16 x 64 each).
  BfPack aqw[2], aqr[2];
#pragma unroll
  for (int kc = 0; kc < 2; ++kc) {
    size_t base = (size_t)((i0 + lcol) * BSZ + b) * DM + n * DH + 32 * kc + 8 * lh;
    const unsigned short* qp = qwbf + base;
    aqw[kc].q[0] = *(const u32x4*)qp;  aqw[kc].q[1] = *(const u32x4*)(qp + 16);
    const unsigned short* rp = qrbf + base;
    aqr[kc].q[0] = *(const u32x4*)rp;  aqr[kc].q[1] = *(const u32x4*)(rp + 16);
  }

  // Per-wave TDM issue for key-step jt into LDS buffer bi.
  auto issue_tdm = [&](int jt, int bi) {
    int j0 = jt * 32;
    if (w == 0) {
      tdm_load_2d(ldsoff(&kbuf[bi][0][0]),
                  kbf + ((size_t)(j0 * BSZ + b) * DM + n * DH),
                  /*tile0=*/64, /*tile1=*/32, /*stride=*/BSZ * DM);
    } else if (w == 1) {
      tdm_load_2d(ldsoff(&vbuf[bi][0][0]),
                  vTbf + ((size_t)((b * NH + n) * DH) * KLEN + j0),
                  /*tile0=*/32, /*tile1=*/64, /*stride=*/KLEN);
    } else {
      int jr0 = j0 + QLEN - i0base - 63 + (w - 2) * 48;   // window rows [jr0, jr0+48)
      tdm_load_2d(ldsoff(&krbuf[bi][(w - 2) * 48][0]),
                  krbf + ((size_t)(jr0 * BSZ + b) * DM + n * DH),
                  /*tile0=*/64, /*tile1=*/48, /*stride=*/BSZ * DM);
    }
  };

  f32x8 zero = {};
  f32x8 acc[4] = {zero, zero, zero, zero};
  float m_run[8], s_run[8];
#pragma unroll
  for (int v = 0; v < 8; ++v) { m_run[v] = -3.0e38f; s_run[v] = 0.f; }

  issue_tdm(0, 0);
  __builtin_amdgcn_s_wait_tensorcnt(0);
  __syncthreads();

  for (int jt = 0; jt < KLEN / 32; ++jt) {
    int bi = jt & 1;
    if (jt + 1 < KLEN / 32) issue_tdm(jt + 1, bi ^ 1);   // prefetch next step
    int j0 = jt * 32;

    // ---- scores: ac from kbuf, bd_raw window tiles from krbuf ----
    f32x8 cc0 = zero, cc1 = zero, T0 = zero, T1 = zero, T2 = zero;
    int kr0 = 48 - 16 * w;                               // wave's window base row
#pragma unroll
    for (int kc = 0; kc < 2; ++kc) {
      BfPack bm;
      const unsigned short* p;
      p = &kbuf[bi][lcol][32 * kc + 8 * lh];
      bm.q[0] = *(const u32x4*)p;  bm.q[1] = *(const u32x4*)(p + 16);
      cc0 = wmma_bf16(aqw[kc], bm, cc0);
      p = &kbuf[bi][16 + lcol][32 * kc + 8 * lh];
      bm.q[0] = *(const u32x4*)p;  bm.q[1] = *(const u32x4*)(p + 16);
      cc1 = wmma_bf16(aqw[kc], bm, cc1);
      p = &krbuf[bi][kr0 + lcol][32 * kc + 8 * lh];
      bm.q[0] = *(const u32x4*)p;  bm.q[1] = *(const u32x4*)(p + 16);
      T0 = wmma_bf16(aqr[kc], bm, T0);
      p = &krbuf[bi][kr0 + 16 + lcol][32 * kc + 8 * lh];
      bm.q[0] = *(const u32x4*)p;  bm.q[1] = *(const u32x4*)(p + 16);
      T1 = wmma_bf16(aqr[kc], bm, T1);
      p = &krbuf[bi][kr0 + 32 + lcol][32 * kc + 8 * lh];
      bm.q[0] = *(const u32x4*)p;  bm.q[1] = *(const u32x4*)(p + 16);
      T2 = wmma_bf16(aqr[kc], bm, T2);
    }

    // ---- rel_shift extraction + ef + mask + online softmax ----
#pragma unroll
    for (int v = 0; v < 8; ++v) {
      int di = v + 8 * lh;
      int c_lin = lcol + 15 - di;                // 0..30 within 48-row window
      int srcl  = (c_lin & 15) | (lane & 16);
      float sh0 = __shfl(T0[v], srcl, 32);
      float sh1 = __shfl(T1[v], srcl, 32);
      float sh2 = __shfl(T2[v], srcl, 32);
      float bd0 = (c_lin < 16) ? sh0 : sh1;
      float bd1 = (c_lin < 16) ? sh1 : sh2;

      int row = i0 + di;
      const float* ep = e2 + ((size_t)(row * BSZ + b) * NH + n) * 2;
      float e0v = ep[0], e1v = ep[1];
      int jA = j0 + lcol, jB = jA + 16;
      float efA = segmat[((size_t)row * KLEN + jA) * BSZ + b] ? e1v : e0v;
      float efB = segmat[((size_t)row * KLEN + jB) * BSZ + b] ? e1v : e0v;
      float s0 = (jA > row + MLEN) ? -1.0e30f : (cc0[v] + bd0 + efA) * ATT_SCALE;
      float s1 = (jB > row + MLEN) ? -1.0e30f : (cc1[v] + bd1 + efB) * ATT_SCALE;

      float rm = fmaxf(s0, s1);
      rm = fmaxf(rm, __shfl_xor(rm, 1, 32));
      rm = fmaxf(rm, __shfl_xor(rm, 2, 32));
      rm = fmaxf(rm, __shfl_xor(rm, 4, 32));
      rm = fmaxf(rm, __shfl_xor(rm, 8, 32));
      float mn = fmaxf(m_run[v], rm);
      float sc = __expf(m_run[v] - mn);
      float p0 = __expf(s0 - mn);
      float p1 = __expf(s1 - mn);
      float rs = p0 + p1;
      rs += __shfl_xor(rs, 1, 32);
      rs += __shfl_xor(rs, 2, 32);
      rs += __shfl_xor(rs, 4, 32);
      rs += __shfl_xor(rs, 8, 32);
      s_run[v] = s_run[v] * sc + rs;
      m_run[v] = mn;
#pragma unroll
      for (int t = 0; t < 4; ++t) acc[t][v] *= sc;
      plds[w][di][lcol]      = f2bf(p0);
      plds[w][di][lcol + 16] = f2bf(p1);
    }
    __builtin_amdgcn_wave_barrier();

    // ---- PV: A = probs (LDS, C->A relayout), B = V^T tiles from vbuf ----
    BfPack pa;
    const unsigned short* lp = &plds[w][lcol][8 * lh];
    pa.q[0] = *(const u32x4*)lp;
    pa.q[1] = *(const u32x4*)(lp + 16);
#pragma unroll
    for (int t = 0; t < 4; ++t) {
      BfPack bm;
      const unsigned short* vp = &vbuf[bi][16 * t + lcol][8 * lh];
      bm.q[0] = *(const u32x4*)vp;  bm.q[1] = *(const u32x4*)(vp + 16);
      acc[t] = wmma_bf16(pa, bm, acc[t]);
    }

    __builtin_amdgcn_s_wait_tensorcnt(0);      // next buffer landed
    __syncthreads();                            // all waves done with buf bi
  }

  // normalize and store attn_vec as bf16 [(i*2+b), n*64+d]
#pragma unroll
  for (int v = 0; v < 8; ++v) {
    float inv = 1.0f / s_run[v];
    int row = i0 + v + 8 * lh;
    size_t obase = (size_t)(row * BSZ + b) * DM + n * DH + lcol;
#pragma unroll
    for (int t = 0; t < 4; ++t)
      attnbf[obase + 16 * t] = f2bf(acc[t][v] * inv);
  }
}

// --------------------------- residual + layernorm ---------------------------
__global__ __launch_bounds__(256)
void k_ln_residual(const float* __restrict__ o, const float* __restrict__ h,
                   const float* __restrict__ g, const float* __restrict__ be,
                   float* __restrict__ y) {
  int row = blockIdx.x;
  int tid = threadIdx.x;
  __shared__ float red[256];
  const float* orow = o + (size_t)row * DM;
  const float* hrow = h + (size_t)row * DM;
  float s = 0.f;
  for (int c = tid; c < DM; c += 256) s += orow[c] + hrow[c];
  red[tid] = s; __syncthreads();
  for (int st = 128; st > 0; st >>= 1) { if (tid < st) red[tid] += red[tid + st]; __syncthreads(); }
  float mu = red[0] * (1.0f / DM);
  __syncthreads();
  float vs = 0.f;
  for (int c = tid; c < DM; c += 256) { float d = orow[c] + hrow[c] - mu; vs += d * d; }
  red[tid] = vs; __syncthreads();
  for (int st = 128; st > 0; st >>= 1) { if (tid < st) red[tid] += red[tid + st]; __syncthreads(); }
  float rstd = rsqrtf(red[0] * (1.0f / DM) + 1e-12f);
  for (int c = tid; c < DM; c += 256) {
    float x = orow[c] + hrow[c];
    y[(size_t)row * DM + c] = (x - mu) * rstd * g[c] + be[c];
  }
}

// --------------------------------- launcher ---------------------------------
extern "C" void kernel_launch(void* const* d_in, const int* in_sizes, int n_in,
                              void* d_out, int out_size, void* d_ws, size_t ws_size,
                              hipStream_t stream) {
  (void)in_sizes; (void)n_in; (void)out_size; (void)ws_size;
  const float* h    = (const float*)d_in[0];
  const float* r    = (const float*)d_in[1];
  const float* mems = (const float*)d_in[2];
  const float* rwb  = (const float*)d_in[3];
  const float* rrb  = (const float*)d_in[4];
  const float* rsb  = (const float*)d_in[5];
  const float* sege = (const float*)d_in[6];
  /* d_in[7] attn_mask: pure function of indices, recomputed inline */
  const float* wq   = (const float*)d_in[8];
  const float* wk   = (const float*)d_in[9];
  const float* wv   = (const float*)d_in[10];
  const float* wr   = (const float*)d_in[11];
  const float* wo   = (const float*)d_in[12];
  const float* lng  = (const float*)d_in[13];
  const float* lnb  = (const float*)d_in[14];
  const unsigned char* segmat = (const unsigned char*)d_in[15];   // bool -> u8

  const int NC_CAT = KLEN * BSZ * DM;
  const int NC_R   = RLEN * BSZ * DM;
  const int NW     = DM * DM;
  const int NQ     = QLEN * BSZ * DM;
  const int NK     = NC_CAT;
  const int NVT    = BSZ * NH * DH * KLEN;
  const int NEROW  = QLEN * BSZ * NH;

  char* ws = (char*)d_ws;
  size_t off = 0;
  auto alloc = [&](size_t bytes) -> void* {
    void* p = ws + off;
    off += (bytes + 255) & ~(size_t)255;
    return p;
  };
  unsigned short* cat_bf = (unsigned short*)alloc((size_t)NC_CAT * 2);
  unsigned short* r_bf   = (unsigned short*)alloc((size_t)NC_R * 2);
  unsigned short* wtq    = (unsigned short*)alloc((size_t)NW * 2);
  unsigned short* wtk    = (unsigned short*)alloc((size_t)NW * 2);
  unsigned short* wtv    = (unsigned short*)alloc((size_t)NW * 2);
  unsigned short* wtr    = (unsigned short*)alloc((size_t)NW * 2);
  unsigned short* wo_bf  = (unsigned short*)alloc((size_t)NW * 2);
  float*          qf     = (float*)alloc((size_t)NQ * 4);
  float*          kf     = (float*)alloc((size_t)NK * 4);
  float*          vf     = (float*)alloc((size_t)NK * 4);
  float*          krf    = (float*)alloc((size_t)NC_R * 4);
  unsigned short* qw_bf  = (unsigned short*)alloc((size_t)NQ * 2);
  unsigned short* qr_bf  = (unsigned short*)alloc((size_t)NQ * 2);
  unsigned short* k_bf   = (unsigned short*)alloc((size_t)NK * 2);
  unsigned short* kr_bf  = (unsigned short*)alloc(((size_t)NC_R + 2 * DM) * 2); // +pad jr=3072
  unsigned short* vT_bf  = (unsigned short*)alloc((size_t)NVT * 2);
  float*          e2     = (float*)alloc((size_t)NEROW * 2 * 4);
  unsigned short* at_bf  = (unsigned short*)alloc((size_t)NQ * 2);
  float*          outf   = (float*)alloc((size_t)NQ * 4);

  auto blocks = [](int n) { return (n + 255) / 256; };

  // pack / convert
  k_cat_cvt<<<blocks(NC_CAT), 256, 0, stream>>>(mems, h, cat_bf);
  k_cvt<<<blocks(NC_R), 256, 0, stream>>>(r, r_bf, NC_R);
  k_wtrans<<<blocks(NW), 256, 0, stream>>>(wq, wtq);
  k_wtrans<<<blocks(NW), 256, 0, stream>>>(wk, wtk);
  k_wtrans<<<blocks(NW), 256, 0, stream>>>(wv, wtv);
  k_wtrans<<<blocks(NW), 256, 0, stream>>>(wr, wtr);
  k_cvt<<<blocks(NW), 256, 0, stream>>>(wo, wo_bf, NW);

  // projections: 32x64 tile per wave -> (M/32)*(768/64)/4 blocks
  k_gemm_bf16<<<(2048 / 32) * 12 / 4, 128, 0, stream>>>(cat_bf + (size_t)2048 * DM, wtq, qf, 2048, DM, DM);
  k_gemm_bf16<<<(4096 / 32) * 12 / 4, 128, 0, stream>>>(cat_bf, wtk, kf, 4096, DM, DM);
  k_gemm_bf16<<<(4096 / 32) * 12 / 4, 128, 0, stream>>>(cat_bf, wtv, vf, 4096, DM, DM);
  k_gemm_bf16<<<(6144 / 32) * 12 / 4, 128, 0, stream>>>(r_bf, wtr, krf, 6144, DM, DM);

  // derived attention operands
  k_addbias_cvt<<<blocks(NQ), 256, 0, stream>>>(qf, rwb, qw_bf, NQ);
  k_addbias_cvt<<<blocks(NQ), 256, 0, stream>>>(qf, rrb, qr_bf, NQ);
  k_cvt<<<blocks(NK), 256, 0, stream>>>(kf, k_bf, NK);
  k_cvt<<<blocks(NC_R), 256, 0, stream>>>(krf, kr_bf, NC_R);
  k_vtrans<<<blocks(NVT), 256, 0, stream>>>(vf, vT_bf);
  k_eseg<<<blocks(NEROW), 256, 0, stream>>>(qf, rsb, sege, e2);

  // fused attention: block = 4 waves on one (b, n, 64 query rows)
  k_xl_attention<<<BSZ * NH * (QLEN / 64), 128, 0, stream>>>(
      qw_bf, qr_bf, k_bf, kr_bf, vT_bf, e2, segmat, at_bf);

  // output projection (wo natural layout is already B^T here)
  k_gemm_bf16<<<(2048 / 32) * 12 / 4, 128, 0, stream>>>(at_bf, wo_bf, outf, 2048, DM, DM);

  // residual + layernorm -> d_out (f32, [q,b,hm] flat)
  k_ln_residual<<<QLEN * BSZ, 256, 0, stream>>>(outf, h, lng, lnb, (float*)d_out);
}
